// SE3Transformer_53523882442974
// MI455X (gfx1250) — compile-verified, matrix operand order checked by reading, not run on hardware
//
#include <hip/hip_runtime.h>
#include <hip/hip_bf16.h>

typedef __attribute__((ext_vector_type(16))) _Float16 v16h;
typedef __attribute__((ext_vector_type(8)))  _Float16 v8h;
typedef __attribute__((ext_vector_type(8)))  float    v8f;

#define NN 1024   // nodes per batch
#define KE 32     // neighbors per node
#define HSTR 136  // padded LDS row stride (halfs): 272B = 68 dwords -> 4-bank shift/row

__device__ __forceinline__ float gelu_f(float x) {
    return 0.5f * x * (1.0f + erff(x * 0.70710678118654752f));
}

// Sum over each 16-lane half of the wave using DPP16 only (no LDS port traffic).
// Butterfly: xor1 (quad_perm 1,0,3,2), xor2 (quad_perm 2,3,0,1),
// xor7-within-8 (row_half_mirror), xor15-within-16 (row_mirror).
// Every lane ends with its 16-lane group's full sum.
template <int CTRL>
__device__ __forceinline__ float dpp_add(float x) {
    int v = __builtin_amdgcn_update_dpp(0, __float_as_int(x), CTRL, 0xF, 0xF, true);
    return x + __int_as_float(v);
}
__device__ __forceinline__ float red16(float x) {
    x = dpp_add<0xB1>(x);    // quad_perm(1,0,3,2)  : + lane^1
    x = dpp_add<0x4E>(x);    // quad_perm(2,3,0,1)  : + lane^2
    x = dpp_add<0x141>(x);   // row_half_mirror     : + lane^7 (within 8)
    x = dpp_add<0x140>(x);   // row_mirror          : + lane^15 (within 16)
    return x;
}

// Load a 16x32 f16 WMMA A/B fragment from a row-major [16 x K] array (rowStride in halfs).
// Lane l (l&15 = row): chunk0 at k = ((l>>4)<<3), chunk1 at +16. (CDNA5 16-bit A/B layout)
template <typename PT>
__device__ __forceinline__ v16h load_frag(PT base, int rowStride, int lane) {
    const int r  = lane & 15;
    const int ko = (lane >> 4) << 3;
    PT p = base + r * rowStride + ko;
    v8h lo = *(const v8h*)(p);
    v8h hi = *(const v8h*)(p + 16);
    v16h a;
#pragma unroll
    for (int i = 0; i < 8; ++i) { a[i] = lo[i]; a[i + 8] = hi[i]; }
    return a;
}

// ---------------- prep: pack weights to f16, transposed [n][k] ----------------
__global__ void se3_prep_w(const float* __restrict__ W2,
                           const float* __restrict__ W300, const float* __restrict__ W301,
                           const float* __restrict__ W310, const float* __restrict__ W311,
                           _Float16* __restrict__ W2T, _Float16* __restrict__ W3T) {
    int idx = blockIdx.x * blockDim.x + threadIdx.x;
    if (idx >= 262144) return;
    if (idx < 65536) {                       // W2T[t][o][k] = W2[t][k][o]
        int t = idx >> 14, r = idx & 16383, o = r >> 7, k = r & 127;
        W2T[idx] = (_Float16)W2[(t * 128 + k) * 128 + o];
    } else {                                 // W3T[nglob][k]
        int j = idx - 65536, ng = j >> 7, k = j & 127;
        float v;
        if      (ng < 256) v = W300[k * 256 + ng];
        else if (ng < 512) v = W301[k * 256 + (ng - 256)];
        else if (ng < 768) v = W310[k * 256 + (ng - 512)];
        else               v = W311[k * 768 + (ng - 768)];
        W3T[j] = (_Float16)v;
    }
}

// layer-1 LN stats: input is affine in d => mu/var are quadratic in d
__global__ void se3_prep_stats(const float* __restrict__ W1, const float* __restrict__ b1,
                               float* __restrict__ stats) {
    __shared__ float sw[128], sb[128], sw2[128], swb[128], sb2[128];
    int t = blockIdx.x, c = threadIdx.x;
    float w = W1[t * 128 + c], bb = b1[t * 128 + c];
    sw[c] = w; sb[c] = bb; sw2[c] = w * w; swb[c] = w * bb; sb2[c] = bb * bb;
    __syncthreads();
    if (c == 0) {
        float a = 0, b2 = 0, cw = 0, dw = 0, e2 = 0;
        for (int i = 0; i < 128; ++i) { a += sw[i]; b2 += sb[i]; cw += sw2[i]; dw += swb[i]; e2 += sb2[i]; }
        float mW = a * (1.0f / 128.0f), mb = b2 * (1.0f / 128.0f);
        stats[t * 8 + 0] = mW;
        stats[t * 8 + 1] = mb;
        stats[t * 8 + 2] = cw * (1.0f / 128.0f) - mW * mW;   // var(W1)
        stats[t * 8 + 3] = dw * (1.0f / 128.0f) - mW * mb;   // cov(W1,b1)
        stats[t * 8 + 4] = e2 * (1.0f / 128.0f) - mb * mb;   // var(b1)
    }
}

// ---------------- fused main kernel: one block per (b,n) node ----------------
__global__ __launch_bounds__(128) void se3_main(
    const float* __restrict__ x0, const float* __restrict__ x1,
    const int* __restrict__ nbr_idx, const unsigned char* __restrict__ nbr_mask,
    const float* __restrict__ rel_dist,
    const float* __restrict__ basis00, const float* __restrict__ basis01,
    const float* __restrict__ basis10, const float* __restrict__ basis11,
    const float* __restrict__ rp_W1, const float* __restrict__ rp_b1,
    const float* __restrict__ rp_g1, const float* __restrict__ rp_be1,
    const float* __restrict__ rp_b2, const float* __restrict__ rp_g2,
    const float* __restrict__ rp_be2,
    const float* __restrict__ b300, const float* __restrict__ b301,
    const float* __restrict__ b310, const float* __restrict__ b311,
    const float* __restrict__ w_si0, const float* __restrict__ w_si1,
    const _Float16* __restrict__ W2T, const _Float16* __restrict__ W3T,
    const float* __restrict__ stats,
    float* __restrict__ out)
{
    __shared__ int   s_idx[32];
    __shared__ float s_d[32], s_fm[32], s_b00[32], s_b01[96];
    __shared__ float s_xg0[32 * 16];
    __shared__ float s_xg1[32 * 16 * 3];
    __shared__ float s_v10[32 * 16];
    __shared__ float s_T[32 * 48 * 3];
    __shared__ __align__(16) _Float16 s_h1[32 * HSTR];
    __shared__ __align__(16) _Float16 s_h2[32 * HSTR];
    __shared__ float s_sum[32], s_sq[32], s_mu[32], s_rs[32];
    __shared__ float s_pool0[16], s_pool1[48];
    __shared__ float s_den;

    const int tid  = threadIdx.x;
    const int lane = tid & 31;
    const int wave = tid >> 5;
    const int li   = lane & 15;
    const int eoff = (lane >> 4) << 3;
    const int bid  = blockIdx.x;
    const int b    = bid >> 10;
    const int n    = bid & (NN - 1);
    const long nodeBase = (long)(b * NN + n);
    const long edge0    = nodeBase * KE;

    // ---- setup: per-edge scalars ----
    if (tid < 32) {
        int e = tid;
        s_idx[e] = nbr_idx[edge0 + e];
        s_d[e]   = rel_dist[edge0 + e];
        s_fm[e]  = nbr_mask[edge0 + e] ? 1.0f : 0.0f;
        s_b00[e] = basis00[edge0 + e];
        for (int p = 0; p < 3; ++p) s_b01[e * 3 + p] = basis01[(edge0 + e) * 3 + p];
    }
    if (tid < 16) s_pool0[tid] = 0.0f;
    if (tid < 48) s_pool1[tid] = 0.0f;
    __syncthreads();
    if (tid == 0) {
        float c = 0; for (int e = 0; e < 32; ++e) c += s_fm[e];
        s_den = fmaxf(c, 1.0f);
    }
    // ---- gather neighbor features (2048 elements) ----
    for (int it = 0; it < 16; ++it) {
        int flat = tid + it * 128;
        int e = flat >> 6, r = flat & 63;
        int node = s_idx[e];
        if (r < 16) {
            s_xg0[e * 16 + r] = x0[((long)b * NN + node) * 16 + r];
        } else {
            int r2 = r - 16, i = r2 / 3, q = r2 % 3;
            s_xg1[(e * 16 + i) * 3 + q] = x1[(((long)b * NN + node) * 16 + i) * 3 + q];
        }
    }
    __syncthreads();
    if (tid < 32) s_fm[tid] *= (1.0f / s_den);           // fold mask/denom
    // v10[e][i] = sum_q basis10[e,q] * xg1[e,i,q]
    for (int it = 0; it < 4; ++it) {
        int flat = tid + it * 128;
        int e = flat >> 4, i = flat & 15;
        float s = 0;
        for (int q = 0; q < 3; ++q) s += basis10[(edge0 + e) * 3 + q] * s_xg1[(e * 16 + i) * 3 + q];
        s_v10[e * 16 + i] = s;
    }
    // T[e][i*3+f][p] = sum_q basis11[e,p,q,f] * xg1[e,i,q]
    for (int it = 0; it < 36; ++it) {
        int flat = tid + it * 128;                        // < 4608
        int e = flat / 144, r = flat % 144, ifl = r / 3, p = r % 3, i = ifl / 3, f = ifl % 3;
        float s = 0;
        for (int q = 0; q < 3; ++q)
            s += basis11[(edge0 + e) * 27 + p * 9 + q * 3 + f] * s_xg1[(e * 16 + i) * 3 + q];
        s_T[(e * 48 + ifl) * 3 + p] = s;
    }
    __syncthreads();

    // ---- radial MLP per (d_in,d_out) pair t, fully fused ----
    for (int t = 0; t < 4; ++t) {
        // layer 1: affine-in-d LN (analytic) + gelu -> f16 LDS
        const float mW = stats[t * 8 + 0], mb = stats[t * 8 + 1];
        const float vW = stats[t * 8 + 2], cWb = stats[t * 8 + 3], vb = stats[t * 8 + 4];
        const float w1c  = rp_W1[t * 128 + tid];
        const float b1c  = rp_b1[t * 128 + tid];
        const float g1c  = rp_g1[t * 128 + tid];
        const float be1c = rp_be1[t * 128 + tid];
        for (int e = 0; e < 32; ++e) {
            float de  = s_d[e];
            float mu  = de * mW + mb;
            float var = de * de * vW + 2.0f * de * cWb + vb;
            float rs  = rsqrtf(var + 1e-5f);
            float xv  = (de * w1c + b1c - mu) * rs * g1c + be1c;
            s_h1[e * HSTR + tid] = (_Float16)gelu_f(xv);
        }
        if (tid < 32) { s_sum[tid] = 0.0f; s_sq[tid] = 0.0f; }
        __syncthreads();

        // layer 2 GEMM: [32x128] @ [128x128], 16 tiles, 4 per wave
        v8f acc[4];
#pragma unroll
        for (int j = 0; j < 4; ++j) {
            int tile = wave * 4 + j;
            int m = tile >> 3, nn2 = tile & 7;
            v8f c = {};
#pragma unroll
            for (int kk = 0; kk < 4; ++kk) {
                v16h a  = load_frag(&s_h1[m * 16 * HSTR + kk * 32], HSTR, lane);
                v16h bf = load_frag(&W2T[(t * 128 + nn2 * 16) * 128 + kk * 32], 128, lane);
                c = __builtin_amdgcn_wmma_f32_16x16x32_f16(false, a, false, bf, (short)0, c, false, false);
            }
            float bb = rp_b2[t * 128 + nn2 * 16 + li];
#pragma unroll
            for (int r = 0; r < 8; ++r) {
                float v = c[r] + bb;
                c[r] = v;
                float sv = red16(v);
                float sq = red16(v * v);
                if (li == 0) {
                    int e = m * 16 + r + eoff;
                    atomicAdd(&s_sum[e], sv);
                    atomicAdd(&s_sq[e], sq);
                }
            }
            acc[j] = c;
        }
        __syncthreads();
        if (tid < 32) {
            float mean = s_sum[tid] * (1.0f / 128.0f);
            float var  = s_sq[tid] * (1.0f / 128.0f) - mean * mean;
            s_mu[tid] = mean;
            s_rs[tid] = rsqrtf(var + 1e-5f);
        }
        __syncthreads();
        // apply LN + gelu, store h2 f16
#pragma unroll
        for (int j = 0; j < 4; ++j) {
            int tile = wave * 4 + j;
            int m = tile >> 3, nn2 = tile & 7;
            float g2v  = rp_g2[t * 128 + nn2 * 16 + li];
            float be2v = rp_be2[t * 128 + nn2 * 16 + li];
            v8f c = acc[j];
#pragma unroll
            for (int r = 0; r < 8; ++r) {
                int e = m * 16 + r + eoff;
                float xv = (c[r] - s_mu[e]) * s_rs[e] * g2v + be2v;
                s_h2[e * HSTR + nn2 * 16 + li] = (_Float16)gelu_f(xv);
            }
        }
        __syncthreads();

        // layer 3 GEMM [32x128] @ [128 x Nt] + in-register basis contraction + pooling
        const int cnt  = (t < 3) ? 8 : 24;
        const int segb = (t == 0) ? 0 : (t == 1) ? 256 : (t == 2) ? 512 : 768;
        const float* b3p = (t == 0) ? b300 : (t == 1) ? b301 : (t == 2) ? b310 : b311;
        for (int j = 0; j < cnt; ++j) {
            int tile = wave * cnt + j;
            int m = tile & 1, nn2 = tile >> 1;
            v8f c = {};
#pragma unroll
            for (int kk = 0; kk < 4; ++kk) {
                v16h a  = load_frag(&s_h2[m * 16 * HSTR + kk * 32], HSTR, lane);
                v16h bf = load_frag(&W3T[(segb + nn2 * 16) * 128 + kk * 32], 128, lane);
                c = __builtin_amdgcn_wmma_f32_16x16x32_f16(false, a, false, bf, (short)0, c, false, false);
            }
            float bb = b3p[nn2 * 16 + li];
            if (t == 0) {             // out0[o=nn2] += b00[e] * sum_i R[o,i]*xg0[e,i]
                float a0 = 0;
#pragma unroll
                for (int r = 0; r < 8; ++r) {
                    int e = m * 16 + r + eoff;
                    float s = red16((c[r] + bb) * s_xg0[e * 16 + li]);
                    a0 += s * s_b00[e] * s_fm[e];
                }
                if (li == 0) atomicAdd(&s_pool0[nn2], a0);
            } else if (t == 1) {      // out1[o=nn2,p] += b01[e,p] * sum_i R[o,i]*xg0[e,i]
                float ap[3] = {0, 0, 0};
#pragma unroll
                for (int r = 0; r < 8; ++r) {
                    int e = m * 16 + r + eoff;
                    float s = red16((c[r] + bb) * s_xg0[e * 16 + li]) * s_fm[e];
#pragma unroll
                    for (int p = 0; p < 3; ++p) ap[p] += s * s_b01[e * 3 + p];
                }
                if (li == 0)
                    for (int p = 0; p < 3; ++p) atomicAdd(&s_pool1[nn2 * 3 + p], ap[p]);
            } else if (t == 2) {      // out0[o=nn2] += sum_i R[o,i]*v10[e,i]
                float a0 = 0;
#pragma unroll
                for (int r = 0; r < 8; ++r) {
                    int e = m * 16 + r + eoff;
                    float s = red16((c[r] + bb) * s_v10[e * 16 + li]);
                    a0 += s * s_fm[e];
                }
                if (li == 0) atomicAdd(&s_pool0[nn2], a0);
            } else {                  // out1[o,p] += sum_{i,f} R[o,i,f]*T[e,i,f,p]
                int o   = nn2 / 3;
                int ifl = (nn2 % 3) * 16 + li;
                float ap[3] = {0, 0, 0};
#pragma unroll
                for (int r = 0; r < 8; ++r) {
                    int e = m * 16 + r + eoff;
                    float val = c[r] + bb;
#pragma unroll
                    for (int p = 0; p < 3; ++p)
                        ap[p] += red16(val * s_T[(e * 48 + ifl) * 3 + p]) * s_fm[e];
                }
                if (li == 0)
                    for (int p = 0; p < 3; ++p) atomicAdd(&s_pool1[o * 3 + p], ap[p]);
            }
        }
        __syncthreads();
    }

    // ---- self-interaction residual + write outputs ----
    if (tid < 16) {
        float si = 0;
#pragma unroll
        for (int i = 0; i < 16; ++i) si += x0[nodeBase * 16 + i] * w_si0[i * 16 + tid];
        out[nodeBase * 16 + tid] = s_pool0[tid] + si;
    }
    if (tid >= 32 && tid < 80) {
        int r = tid - 32;
        int co = r / 3, p = r % 3;
        float si = 0;
#pragma unroll
        for (int i = 0; i < 16; ++i) si += x1[(nodeBase * 16 + i) * 3 + p] * w_si1[i * 16 + co];
        out[2 * NN * 16 + (nodeBase * 16 + co) * 3 + p] = s_pool1[r] + si;
    }
}

extern "C" void kernel_launch(void* const* d_in, const int* in_sizes, int n_in,
                              void* d_out, int out_size, void* d_ws, size_t ws_size,
                              hipStream_t stream) {
    const float* x0       = (const float*)d_in[0];
    const float* x1       = (const float*)d_in[1];
    const int*   nbr_idx  = (const int*)d_in[2];
    const unsigned char* nbr_mask = (const unsigned char*)d_in[3];
    const float* rel_dist = (const float*)d_in[4];
    const float* b00      = (const float*)d_in[5];
    const float* b01      = (const float*)d_in[6];
    const float* b10      = (const float*)d_in[7];
    const float* b11      = (const float*)d_in[8];
    const float* rp_W1    = (const float*)d_in[9];
    const float* rp_b1    = (const float*)d_in[10];
    const float* rp_g1    = (const float*)d_in[11];
    const float* rp_be1   = (const float*)d_in[12];
    const float* rp_W2    = (const float*)d_in[13];
    const float* rp_b2    = (const float*)d_in[14];
    const float* rp_g2    = (const float*)d_in[15];
    const float* rp_be2   = (const float*)d_in[16];
    const float* W300     = (const float*)d_in[17];
    const float* b300     = (const float*)d_in[18];
    const float* W301     = (const float*)d_in[19];
    const float* b301     = (const float*)d_in[20];
    const float* W310     = (const float*)d_in[21];
    const float* b310     = (const float*)d_in[22];
    const float* W311     = (const float*)d_in[23];
    const float* b311     = (const float*)d_in[24];
    const float* w_si0    = (const float*)d_in[25];
    const float* w_si1    = (const float*)d_in[26];

    char* ws = (char*)d_ws;
    _Float16* W2T  = (_Float16*)ws;                       // 4*128*128 halfs = 128 KB
    _Float16* W3T  = (_Float16*)(ws + 131072);            // 1536*128 halfs = 384 KB
    float*    stats = (float*)(ws + 131072 + 393216);     // 32 floats

    se3_prep_w<<<1024, 256, 0, stream>>>(rp_W2, W300, W301, W310, W311, W2T, W3T);
    se3_prep_stats<<<4, 128, 0, stream>>>(rp_W1, rp_b1, stats);
    se3_main<<<2 * NN, 128, 0, stream>>>(x0, x1, nbr_idx, nbr_mask, rel_dist,
        b00, b01, b10, b11, rp_W1, rp_b1, rp_g1, rp_be1, rp_b2, rp_g2, rp_be2,
        b300, b301, b310, b311, w_si0, w_si1, W2T, W3T, stats, (float*)d_out);
}
